// DEACA_attention_61761629716674
// MI455X (gfx1250) — matrix operand order repaired and checked by reading, not compiled, not run on hardware
//
#include <hip/hip_runtime.h>

// ---------------------------------------------------------------------------
// Types / bf16 helpers
// ---------------------------------------------------------------------------
typedef __attribute__((ext_vector_type(16))) __bf16          v16bf;
typedef __attribute__((ext_vector_type(16))) unsigned short  v16u;
typedef __attribute__((ext_vector_type(8)))  unsigned short  v8u;
typedef __attribute__((ext_vector_type(8)))  float           v8f;

static __device__ __forceinline__ unsigned short f2bf(float f) {
    unsigned u = __builtin_bit_cast(unsigned, f);
    u += 0x7fffu + ((u >> 16) & 1u);          // round-to-nearest-even
    return (unsigned short)(u >> 16);
}
static __device__ __forceinline__ float bf2f(unsigned short h) {
    return __builtin_bit_cast(float, (unsigned)h << 16);
}
static __device__ __forceinline__ unsigned pack2(float lo, float hi) {
    return (unsigned)f2bf(lo) | ((unsigned)f2bf(hi) << 16);
}

// ---------------------------------------------------------------------------
// WMMA fragment loaders (per CDNA5 ISA 7.12.2 layouts, wave32).
// All bases/strides are 16-byte aligned so these lower to ds_load_b128.
// ---------------------------------------------------------------------------
// A 16x32 (MxK), source row-major [m][k] in LDS, row stride `ld` (mult of 8)
static __device__ __forceinline__ v16bf load_a_lds(const unsigned short* base, int ld, int lane) {
    const int half = lane >> 4, m = lane & 15;
    const unsigned short* p = base + m * ld + half * 8;
    v8u lo = *(const v8u*)p;          // k = half*8 + 0..7
    v8u hi = *(const v8u*)(p + 16);   // k = 16 + half*8 + 0..7
    v16u t = __builtin_shufflevector(lo, hi, 0,1,2,3,4,5,6,7,8,9,10,11,12,13,14,15);
    return __builtin_bit_cast(v16bf, t);
}
// B 32x16 (KxN) where source is stored [n][k] row-major (row stride mult of 8)
static __device__ __forceinline__ v16bf load_b_lds_nk(const unsigned short* base, int ld, int lane) {
    const int half = lane >> 4, n = lane & 15;
    const unsigned short* p = base + n * ld + 8 * half;  // k = half*8 + 0..7, 16 + half*8 + 0..7
    v8u lo = *(const v8u*)p;
    v8u hi = *(const v8u*)(p + 16);
    v16u t = __builtin_shufflevector(lo, hi, 0,1,2,3,4,5,6,7,8,9,10,11,12,13,14,15);
    return __builtin_bit_cast(v16bf, t);
}
static __device__ __forceinline__ v8f wmma_bf16(v16bf a, v16bf b, v8f c) {
    return __builtin_amdgcn_wmma_f32_16x16x32_bf16(false, a, false, b, (short)0, c, false, false);
}

// ---------------------------------------------------------------------------
// Problem constants
// ---------------------------------------------------------------------------
#define C_DIM 384
#define NH    6
#define HD    64
#define B_SZ  8
#define T_SZ  128
#define H_SZ  128
#define W_SZ  128

// ---------------------------------------------------------------------------
// Generic bf16 WMMA GEMM:  out[m,n] = (sum_k A[m,k]*W[n,k]) * g[n] + b[n]
// A fp32 [M,K] row-major, W fp32 [N,K] row-major (converted to bf16 on stage)
// MODE 0: fp32 out[m*N+n]
// MODE 1: bf16 head layout: out[(((m>>7)*6 + (n>>6))*128 + (m&127))*64 + (n&63)]
// MODE 2: bf16 out[m*N+n]
// MODE 3: fp32 out[((m&127)*8 + (m>>7))*N + n]   (final [T,B,C] transpose)
// ---------------------------------------------------------------------------
#define GBM 128
#define GBN 64
#define GBK 32
#define GLD 40  // padded, 16B-aligned LDS row stride

template <int MODE>
__global__ __launch_bounds__(256) void gemm_kernel(
    const float* __restrict__ A, const float* __restrict__ W,
    const float* __restrict__ gv, const float* __restrict__ bv,
    float* __restrict__ outf, unsigned short* __restrict__ outb,
    int M, int N, int K)
{
    __shared__ __align__(16) unsigned short As[GBM * GLD];
    __shared__ __align__(16) unsigned short Bs[GBN * GLD];
    const int tid = threadIdx.x, lane = tid & 31, wave = tid >> 5;
    const int wm = wave >> 1, wn = wave & 1;
    const int m0 = blockIdx.x * GBM, n0 = blockIdx.y * GBN;

    v8f acc[2][2] = {};

    for (int k0 = 0; k0 < K; k0 += GBK) {
        {   // stage A tile 128x32: 16 fp32 per thread -> 2x uint4 (8 bf16) stores
            int r = tid >> 1, cs = (tid & 1) * 16;
            const float4* s4 = (const float4*)(A + (size_t)(m0 + r) * K + (k0 + cs));
            float4 fa = s4[0], fb = s4[1], fc = s4[2], fd = s4[3];
            uint4 o0, o1;
            o0.x = pack2(fa.x, fa.y); o0.y = pack2(fa.z, fa.w);
            o0.z = pack2(fb.x, fb.y); o0.w = pack2(fb.z, fb.w);
            o1.x = pack2(fc.x, fc.y); o1.y = pack2(fc.z, fc.w);
            o1.z = pack2(fd.x, fd.y); o1.w = pack2(fd.z, fd.w);
            uint4* dst = (uint4*)(As + r * GLD + cs);
            dst[0] = o0; dst[1] = o1;
        }
        {   // stage B tile 64x32: 8 fp32 per thread -> 1x uint4 store
            int r = tid >> 2, cs = (tid & 3) * 8;
            const float4* s4 = (const float4*)(W + (size_t)(n0 + r) * K + (k0 + cs));
            float4 fa = s4[0], fb = s4[1];
            uint4 o0;
            o0.x = pack2(fa.x, fa.y); o0.y = pack2(fa.z, fa.w);
            o0.z = pack2(fb.x, fb.y); o0.w = pack2(fb.z, fb.w);
            *(uint4*)(Bs + r * GLD + cs) = o0;
        }
        __syncthreads();
        v16bf a0 = load_a_lds(As + (wm * 32) * GLD, GLD, lane);
        v16bf a1 = load_a_lds(As + (wm * 32 + 16) * GLD, GLD, lane);
        v16bf b0 = load_b_lds_nk(Bs + (wn * 32) * GLD, GLD, lane);
        v16bf b1 = load_b_lds_nk(Bs + (wn * 32 + 16) * GLD, GLD, lane);
        acc[0][0] = wmma_bf16(a0, b0, acc[0][0]);
        acc[0][1] = wmma_bf16(a0, b1, acc[0][1]);
        acc[1][0] = wmma_bf16(a1, b0, acc[1][0]);
        acc[1][1] = wmma_bf16(a1, b1, acc[1][1]);
        __syncthreads();
    }

    const int half = lane >> 4, nl = lane & 15;
#pragma unroll
    for (int i = 0; i < 2; ++i)
#pragma unroll
        for (int j = 0; j < 2; ++j) {
            int col = n0 + wn * 32 + j * 16 + nl;
            float gg = gv[col], bb = bv[col];
#pragma unroll
            for (int r = 0; r < 8; ++r) {
                int row = m0 + wm * 32 + i * 16 + r + 8 * half;
                float val = acc[i][j][r] * gg + bb;
                if (MODE == 0) {
                    outf[(size_t)row * N + col] = val;
                } else if (MODE == 1) {
                    int b = row >> 7, t = row & 127, hidx = col >> 6, e = col & 63;
                    outb[(((size_t)(b * NH + hidx)) * 128 + t) * HD + e] = f2bf(val);
                } else if (MODE == 2) {
                    outb[(size_t)row * N + col] = f2bf(val);
                } else {
                    int b = row >> 7, t = row & 127;
                    outf[((size_t)t * B_SZ + b) * N + col] = val;
                }
            }
        }
}

// ---------------------------------------------------------------------------
// Spatial means of k_row / k_col (mean commutes with the 1x1 conv + affine)
// ---------------------------------------------------------------------------
__global__ __launch_bounds__(384) void mean_w_kernel(const float* __restrict__ in, float* __restrict__ out) {
    int bh = blockIdx.x, c = threadIdx.x;                  // in [b,h,w,c] -> out [bh,c]
    const float* p = in + (size_t)bh * W_SZ * C_DIM + c;
    float s = 0.f;
    for (int w = 0; w < W_SZ; ++w) s += p[(size_t)w * C_DIM];
    out[(size_t)bh * C_DIM + c] = s * (1.f / 128.f);
}
__global__ __launch_bounds__(384) void mean_h_kernel(const float* __restrict__ in, float* __restrict__ out) {
    int bw = blockIdx.x, b = bw >> 7, w = bw & 127, c = threadIdx.x;  // -> out [bw,c]
    const float* p = in + (((size_t)b * H_SZ) * W_SZ + w) * C_DIM + c;
    float s = 0.f;
    for (int h = 0; h < H_SZ; ++h) s += p[(size_t)h * W_SZ * C_DIM];
    out[(size_t)bw * C_DIM + c] = s * (1.f / 128.f);
}

// ---------------------------------------------------------------------------
// Scrambled-reshape reductions of vv (bf16 [bhw, c]) into v_row / v_col / vm.
// v4[n,h',w',e] = vv[b, 3h'+(u>>7), u&127, w], u=3w'+q,
//   m=n*64+e, q=m>>10, t=m&1023, b=t>>7, w=t&127.
// ---------------------------------------------------------------------------
__global__ __launch_bounds__(128) void vrow_kernel(const unsigned short* __restrict__ vv,
                                                   unsigned short* __restrict__ vrow,
                                                   float* __restrict__ vm) {
    int n = blockIdx.x >> 6, e = blockIdx.x & 63, hp = threadIdx.x;
    int m = n * 64 + e, q = m >> 10, t = m & 1023, b = t >> 7, w = t & 127;
    float s = 0.f;
    for (int i = 0; i < 128; ++i) {
        int u = 3 * i + q, c = 3 * hp + (u >> 7), h = u & 127;
        s += bf2f(vv[(((size_t)(b * H_SZ + h)) * W_SZ + w) * C_DIM + c]);
    }
    s *= (1.f / 128.f);
    vrow[(((size_t)n) * 128 + hp) * HD + e] = f2bf(s);
    __shared__ float red[128];
    red[hp] = s;
    __syncthreads();
    for (int off = 64; off > 0; off >>= 1) { if (hp < off) red[hp] += red[hp + off]; __syncthreads(); }
    if (hp == 0) vm[n * 64 + e] = red[0] * (1.f / 128.f);
}
__global__ __launch_bounds__(128) void vcol_kernel(const unsigned short* __restrict__ vv,
                                                   unsigned short* __restrict__ vcol) {
    int n = blockIdx.x >> 6, e = blockIdx.x & 63, wp = threadIdx.x;
    int m = n * 64 + e, q = m >> 10, t = m & 1023, b = t >> 7, w = t & 127;
    int u = 3 * wp + q, d = u >> 7, h = u & 127;
    const unsigned short* p = vv + (((size_t)(b * H_SZ + h)) * W_SZ + w) * C_DIM + d;
    float s = 0.f;
    for (int j = 0; j < 128; ++j) s += bf2f(p[3 * j]);
    vcol[(((size_t)n) * 128 + wp) * HD + e] = f2bf(s * (1.f / 128.f));
}

// ---------------------------------------------------------------------------
// SE-style gate: v_avg = sigmoid(W_pw . vm * g + b); scale kr/kc heads in place
// ---------------------------------------------------------------------------
__global__ __launch_bounds__(64) void gate_kernel(const float* __restrict__ vm,
                                                  const float* __restrict__ Wpw,
                                                  const float* __restrict__ gpw,
                                                  const float* __restrict__ bpw,
                                                  unsigned short* __restrict__ krh,
                                                  unsigned short* __restrict__ kch) {
    int n = blockIdx.x, o = threadIdx.x;
    const float* wr = Wpw + (size_t)o * HD;
    const float* vn = vm + (size_t)n * HD;
    float dot = 0.f;
    for (int e = 0; e < HD; ++e) dot += wr[e] * vn[e];
    float va = 1.f / (1.f + __expf(-(dot * gpw[o] + bpw[o])));
    unsigned short* kr = krh + (size_t)n * 128 * HD + o;
    unsigned short* kc = kch + (size_t)n * 128 * HD + o;
    for (int s = 0; s < 128; ++s) {
        kr[(size_t)s * HD] = f2bf(bf2f(kr[(size_t)s * HD]) * va);
        kc[(size_t)s * HD] = f2bf(bf2f(kc[(size_t)s * HD]) * va);
    }
}

// ---------------------------------------------------------------------------
// Per-head attention: scores = q.k^T * scale -> softmax -> P.v, relu, write
// to [bt, c] fp32. One workgroup (8 waves) per head. Q/K staged [s][e],
// V staged transposed [e][s] so the P.V B-fragments are contiguous b128
// loads. P aliases Q+K after the score pass.
// ---------------------------------------------------------------------------
#define LQK 72    // Q/K LDS row stride (16B aligned)
#define LVT 136   // Vt / P LDS row stride (16B aligned)

__global__ __launch_bounds__(256) void attn_kernel(const unsigned short* __restrict__ qh,
                                                   const unsigned short* __restrict__ kh,
                                                   const unsigned short* __restrict__ vh,
                                                   float* __restrict__ xout, float scaling) {
    const int n = blockIdx.x;
    const int tid = threadIdx.x, lane = tid & 31, wave = tid >> 5;
    const int bb = n / NH, coff = (n % NH) * HD;

    __shared__ __align__(16) unsigned short smem[27136];
    unsigned short* Qs = smem;               // 128 x 64, ld LQK (9216)
    unsigned short* Ks = smem + 9216;        // 128 x 64, ld LQK (9216)
    unsigned short* Vt = smem + 18432;       // 64 x 128, ld LVT (8704)
    unsigned short* Ps = smem;               // 128 x 128, ld LVT (17408, aliases Q+K)

    const unsigned short* qp = qh + (size_t)n * 128 * HD;
    const unsigned short* kp = kh + (size_t)n * 128 * HD;
    const unsigned short* vp = vh + (size_t)n * 128 * HD;
    // stage: 1024 chunks of 8 bf16; Q/K copied straight, V transposed
    for (int chunk = tid; chunk < 1024; chunk += 256) {
        int s = chunk >> 3, e0 = (chunk & 7) * 8;
        v8u q8 = *(const v8u*)(qp + s * HD + e0);
        *(v8u*)(Qs + s * LQK + e0) = q8;
        v8u k8 = *(const v8u*)(kp + s * HD + e0);
        *(v8u*)(Ks + s * LQK + e0) = k8;
        v8u v8 = *(const v8u*)(vp + s * HD + e0);
#pragma unroll
        for (int j = 0; j < 8; ++j) Vt[(e0 + j) * LVT + s] = v8[j];
    }
    __syncthreads();

    // scores: wave handles rows [16w,16w+16), 8 col tiles over s, K=64 (2 steps)
    v8f sacc[8] = {};
#pragma unroll
    for (int kk = 0; kk < 2; ++kk) {
        v16bf aF = load_a_lds(Qs + (wave * 16) * LQK + kk * 32, LQK, lane);
#pragma unroll
        for (int j = 0; j < 8; ++j) {
            v16bf bF = load_b_lds_nk(Ks + (j * 16) * LQK + kk * 32, LQK, lane);
            sacc[j] = wmma_bf16(aF, bF, sacc[j]);
        }
    }
    __syncthreads();   // everyone done with Q/K before P overwrites them

    const int half = lane >> 4, nl = lane & 15;
#pragma unroll
    for (int r = 0; r < 8; ++r) {
        float mx = -1e30f;
#pragma unroll
        for (int j = 0; j < 8; ++j) mx = fmaxf(mx, sacc[j][r] * scaling);
        for (int off = 1; off < 16; off <<= 1) mx = fmaxf(mx, __shfl_xor(mx, off, 32));
        float ex[8], sum = 0.f;
#pragma unroll
        for (int j = 0; j < 8; ++j) { ex[j] = __expf(sacc[j][r] * scaling - mx); sum += ex[j]; }
        for (int off = 1; off < 16; off <<= 1) sum += __shfl_xor(sum, off, 32);
        float inv = 1.f / sum;
        int row = wave * 16 + r + 8 * half;
#pragma unroll
        for (int j = 0; j < 8; ++j) Ps[row * LVT + j * 16 + nl] = f2bf(ex[j] * inv);
    }

    // out = P . V : K = 128 (4 steps), 4 col tiles over e; B-frags from Vt [e][s]
    v8f oacc[4] = {};
#pragma unroll
    for (int kk = 0; kk < 4; ++kk) {
        v16bf aF = load_a_lds(Ps + (wave * 16) * LVT + kk * 32, LVT, lane);
#pragma unroll
        for (int j = 0; j < 4; ++j) {
            v16bf bF = load_b_lds_nk(Vt + (j * 16) * LVT + kk * 32, LVT, lane);
            oacc[j] = wmma_bf16(aF, bF, oacc[j]);
        }
    }
#pragma unroll
    for (int j = 0; j < 4; ++j) {
        int e = j * 16 + nl;
#pragma unroll
        for (int r = 0; r < 8; ++r) {
            int t = wave * 16 + r + 8 * half;
            xout[((size_t)bb * T_SZ + t) * C_DIM + coff + e] = fmaxf(oacc[j][r], 0.f);
        }
    }
}

__global__ void addrelu_kernel(const float* __restrict__ a, const float* __restrict__ b,
                               float* __restrict__ o, int nElem) {
    int i = blockIdx.x * blockDim.x + threadIdx.x;
    if (i < nElem) o[i] = fmaxf(a[i] + b[i], 0.f);
}

// ---------------------------------------------------------------------------
// Host-side orchestration
// ---------------------------------------------------------------------------
extern "C" void kernel_launch(void* const* d_in, const int* in_sizes, int n_in,
                              void* d_out, int out_size, void* d_ws, size_t ws_size,
                              hipStream_t stream) {
    (void)in_sizes; (void)n_in; (void)out_size; (void)ws_size;

    const float* q_row = (const float*)d_in[0];
    const float* q_col = (const float*)d_in[1];
    const float* k_row = (const float*)d_in[2];
    const float* k_col = (const float*)d_in[3];
    const float* v_in  = (const float*)d_in[4];
    // params flattened in dict insertion order: (W,g,b) per name
    const float* W_qr = (const float*)d_in[5];  const float* g_qr = (const float*)d_in[6];  const float* b_qr = (const float*)d_in[7];
    const float* W_qc = (const float*)d_in[8];  const float* g_qc = (const float*)d_in[9];  const float* b_qc = (const float*)d_in[10];
    const float* W_kr = (const float*)d_in[11]; const float* g_kr = (const float*)d_in[12]; const float* b_kr = (const float*)d_in[13];
    const float* W_kc = (const float*)d_in[14]; const float* g_kc = (const float*)d_in[15]; const float* b_kc = (const float*)d_in[16];
    const float* W_v  = (const float*)d_in[17]; const float* g_v  = (const float*)d_in[18]; const float* b_v  = (const float*)d_in[19];
    const float* W_pw = (const float*)d_in[20]; const float* g_pw = (const float*)d_in[21]; const float* b_pw = (const float*)d_in[22];
    const float* W_pr = (const float*)d_in[23]; const float* g_pr = (const float*)d_in[24]; const float* b_pr = (const float*)d_in[25];
    const float* W_pc = (const float*)d_in[26]; const float* g_pc = (const float*)d_in[27]; const float* b_pc = (const float*)d_in[28];
    const float* W_p  = (const float*)d_in[29]; const float* g_p  = (const float*)d_in[30]; const float* b_p  = (const float*)d_in[31];

    char* ws = (char*)d_ws;
    const size_t SZ_VV   = (size_t)131072 * 384 * 2;   // bf16 vv [bhw, c]
    const size_t SZ_HEAD = (size_t)48 * 128 * 64 * 2;  // bf16 head tensors
    const size_t SZ_VM   = (size_t)48 * 64 * 4;        // fp32
    const size_t SZ_MAT  = (size_t)1024 * 384 * 4;     // fp32 [bt, c]

    size_t off = 0;
    unsigned short* vv     = (unsigned short*)(ws + off); off += SZ_VV;
    unsigned short* qr_h   = (unsigned short*)(ws + off); off += SZ_HEAD;
    unsigned short* qc_h   = (unsigned short*)(ws + off); off += SZ_HEAD;
    unsigned short* kr_h   = (unsigned short*)(ws + off); off += SZ_HEAD;
    unsigned short* kc_h   = (unsigned short*)(ws + off); off += SZ_HEAD;
    unsigned short* vrow_h = (unsigned short*)(ws + off); off += SZ_HEAD;
    unsigned short* vcol_h = (unsigned short*)(ws + off); off += SZ_HEAD;
    float* vm   = (float*)(ws + off); off += SZ_VM;
    float* krm  = (float*)(ws + off); off += SZ_MAT;
    float* kcm  = (float*)(ws + off); off += SZ_MAT;
    float* xrow = (float*)(ws + off); off += SZ_MAT;
    float* xcol = (float*)(ws + off); off += SZ_MAT;
    float* outr = (float*)(ws + off); off += SZ_MAT;
    float* outc = (float*)(ws + off); off += SZ_MAT;
    float* xsum = (float*)(ws + off); off += SZ_MAT;

    // 1) spatial means of k_row / k_col (mean commutes with 1x1 conv+affine)
    mean_w_kernel<<<1024, 384, 0, stream>>>(k_row, krm);
    mean_h_kernel<<<1024, 384, 0, stream>>>(k_col, kcm);

    // 2) projections (bf16 WMMA GEMMs, BN affine in epilogue)
    dim3 g6(8, 6);
    gemm_kernel<1><<<g6, 256, 0, stream>>>(q_row, W_qr, g_qr, b_qr, nullptr, qr_h, 1024, 384, 384);
    gemm_kernel<1><<<g6, 256, 0, stream>>>(q_col, W_qc, g_qc, b_qc, nullptr, qc_h, 1024, 384, 384);
    gemm_kernel<1><<<g6, 256, 0, stream>>>(krm,   W_kr, g_kr, b_kr, nullptr, kr_h, 1024, 384, 384);
    gemm_kernel<1><<<g6, 256, 0, stream>>>(kcm,   W_kc, g_kc, b_kc, nullptr, kc_h, 1024, 384, 384);
    dim3 gvg(1024, 6);
    gemm_kernel<2><<<gvg, 256, 0, stream>>>(v_in, W_v, g_v, b_v, nullptr, vv, 131072, 384, 384);

    // 3) scrambled-reshape reductions of vv
    vrow_kernel<<<3072, 128, 0, stream>>>(vv, vrow_h, vm);
    vcol_kernel<<<3072, 128, 0, stream>>>(vv, vcol_h);

    // 4) SE gate applied to kr/kc heads
    gate_kernel<<<48, 64, 0, stream>>>(vm, W_pw, g_pw, b_pw, kr_h, kc_h);

    // 5) row/col attention (scale = hd^-0.5 = 0.125), relu'd outputs into [bt,c]
    attn_kernel<<<48, 256, 0, stream>>>(qr_h, kr_h, vrow_h, xrow, 0.125f);
    attn_kernel<<<48, 256, 0, stream>>>(qc_h, kc_h, vcol_h, xcol, 0.125f);

    // 6) output projections
    gemm_kernel<0><<<g6, 256, 0, stream>>>(xrow, W_pr, g_pr, b_pr, outr, nullptr, 1024, 384, 384);
    gemm_kernel<0><<<g6, 256, 0, stream>>>(xcol, W_pc, g_pc, b_pc, outc, nullptr, 1024, 384, 384);
    addrelu_kernel<<<(1024 * 384 + 255) / 256, 256, 0, stream>>>(outr, outc, xsum, 1024 * 384);
    gemm_kernel<3><<<g6, 256, 0, stream>>>(xsum, W_p, g_p, b_p, (float*)d_out, nullptr, 1024, 384, 384);
}
